// transformer_32049045963010
// MI455X (gfx1250) — matrix operand order repaired and checked by reading, not compiled, not run on hardware
//
#include <hip/hip_runtime.h>

// ---------------------------------------------------------------------------
// MI455X / gfx1250 banded multi-head attention (sequential heads).
// Compute-bound (~43 GFLOP, working set < 192MB L2) -> bf16 WMMA path.
// This revision: (a) register-cached A fragments + multi-tile accumulators to
// raise FLOP/byte at the WMMA units, (b) async global->LDS staging of the e
// strip in k_qkv (ASYNCcnt / s_wait_asynccnt CDNA5 path).
// ---------------------------------------------------------------------------

typedef __attribute__((ext_vector_type(16))) __bf16 v16bf;
typedef __attribute__((ext_vector_type(8)))  __bf16 v8bf;
typedef __attribute__((ext_vector_type(4)))  __bf16 v4bf;
typedef __attribute__((ext_vector_type(8)))  float  v8f;

constexpr int   N_  = 4096;
constexpr int   D_  = 1024;
constexpr int   H_  = 8;
constexpr int   DH_ = 128;
constexpr int   CW_ = 256;
constexpr float SCALE_ = 0.08838834764831845f;   // 1/sqrt(128)
constexpr float NEGBIG = -3.0e38f;

__device__ __forceinline__ __bf16 f2bf(float x) {
  union { float f; unsigned u; } v; v.f = x;
  unsigned r = (v.u + 0x7FFFu + ((v.u >> 16) & 1u)) >> 16;  // round-nearest-even
  union { unsigned short s; __bf16 b; } o; o.s = (unsigned short)r;
  return o.b;
}

__device__ __forceinline__ v8f wmma_bf16(v16bf a, v16bf b, v8f c) {
  // D = A(16x32) * B(32x16) + C, f32 accumulate
  return __builtin_amdgcn_wmma_f32_16x16x32_bf16(false, a, false, b, (short)0, c,
                                                 false, false);
}

union V16u { v16bf v; v8bf h[2]; };

// A fragment (16x32 bf16, row-major source, stride in elements).
// Lanes 0-15: row m, K {0..15}; lanes 16-31: row m, K {16..31}.
__device__ __forceinline__ v16bf loadA(const __bf16* tile, int stride, int lane) {
  const int m  = lane & 15;
  const int kh = (lane >> 4) << 4;                 // 0 or 16
  const __bf16* p = tile + m * stride + kh;
  V16u r;
  r.h[0] = *(const v8bf*)p;                        // K = kh..kh+7
  r.h[1] = *(const v8bf*)(p + 8);                  // K = kh+8..kh+15
  return r.v;
}

// B fragment (32x16 bf16). Source holds, for each output column n, 32 contiguous
// K values at colbase + n*cstride (N-major / K-contiguous source).
__device__ __forceinline__ v16bf loadB(const __bf16* colbase, int cstride, int lane) {
  const int n  = lane & 15;
  const int kb = (lane >> 4) << 4;                 // 0 or 16
  return *(const v16bf*)(colbase + n * cstride + kb);
}

// ---------------------------------------------------------------------------
// f32 -> bf16 conversion (4 elements / thread, sizes are exact multiples)
// ---------------------------------------------------------------------------
__global__ void __launch_bounds__(256) k_f32_to_bf16(const float* __restrict__ src,
                                                     __bf16* __restrict__ dst) {
  const int i = (blockIdx.x * 256 + threadIdx.x) * 4;
  const float4 v = *(const float4*)(src + i);
  v4bf o;
  o[0] = f2bf(v.x); o[1] = f2bf(v.y); o[2] = f2bf(v.z); o[3] = f2bf(v.w);
  *(v4bf*)(dst + i) = o;
}

// ---------------------------------------------------------------------------
// QKV projections for one head. One block = one 16-row e strip, staged once
// into LDS with async global->LDS loads, then consumed by 6 waves:
//   wave w: matrix z = w>>1 (q/k/vd), n-tiles (w&1)*4 .. +3  (4 accumulators).
// ---------------------------------------------------------------------------
__global__ void __launch_bounds__(192) k_qkv(const __bf16* __restrict__ ebf,
                                             const __bf16* __restrict__ wq,
                                             const __bf16* __restrict__ wk,
                                             const __bf16* __restrict__ wvd,
                                             __bf16* __restrict__ qb,
                                             __bf16* __restrict__ kb,
                                             __bf16* __restrict__ vdT) {
  __shared__ alignas(16) __bf16 Es[16 * D_];       // 32 KB e strip
  const int j0  = blockIdx.x * 16;
  const int tid = threadIdx.x;

  // ---- async stage: 32768 bytes = 2048 x b128 transfers across 192 threads ----
  {
    const char* g = (const char*)(ebf + j0 * D_);
    for (int ofs = tid * 16; ofs < 16 * D_ * 2; ofs += 192 * 16) {
      unsigned lds = (unsigned)(size_t)((char*)Es + ofs);   // LDS aperture: addr[31:0]
      asm volatile("global_load_async_to_lds_b128 %0, %1, off"
                   :: "v"(lds), "v"(g + ofs) : "memory");
    }
    asm volatile("s_wait_asynccnt 0x0" ::: "memory");
  }
  __syncthreads();

  const int wid  = tid >> 5;                        // 0..5
  const int lane = tid & 31;
  const int z    = wid >> 1;                        // 0=q, 1=k, 2=vd
  const int nb   = (wid & 1) * 4;                   // n-tile base (4 tiles/wave)
  const __bf16* w = (z == 0) ? wq : (z == 1) ? wk : wvd;

  v8f acc[4] = {};
  for (int k0 = 0; k0 < D_; k0 += 32) {
    v16bf a = loadA(Es + k0, D_, lane);             // one A frag, reused 4x
    for (int t = 0; t < 4; ++t) {
      v16bf b = loadB(w + (nb + t) * 16 * D_ + k0, D_, lane);
      acc[t] = wmma_bf16(a, b, acc[t]);
    }
  }

  const int n = lane & 15, mh = (lane >> 4) * 8;    // C layout: (m = mh+r, n)
  for (int t = 0; t < 4; ++t) {
    const int n0 = (nb + t) * 16;
    if (z == 0) {
      for (int r = 0; r < 8; ++r)
        qb[(j0 + mh + r) * DH_ + n0 + n] = f2bf(acc[t][r] * SCALE_);
    } else if (z == 1) {
      for (int r = 0; r < 8; ++r)
        kb[(j0 + mh + r) * DH_ + n0 + n] = f2bf(acc[t][r]);
    } else {
      for (int r = 0; r < 8; ++r)                   // store V transposed [dh][token]
        vdT[(n0 + n) * N_ + (j0 + mh + r)] = f2bf(acc[t][r]);
    }
  }
}

// ---------------------------------------------------------------------------
// Banded attention for a 16-row strip: scores -> softmax (LDS) -> P @ Vd.
// Band columns: [j0-256, j0+271) = 33 tiles of 16 (per-element mask applied).
// q fragments for the whole K=128 are cached in registers (loaded once/wave).
// ---------------------------------------------------------------------------
__global__ void __launch_bounds__(256) k_attn(const __bf16* __restrict__ qb,
                                              const __bf16* __restrict__ kb,
                                              const __bf16* __restrict__ vdT,
                                              __bf16* __restrict__ attn) {
  __shared__ alignas(16) float  S[16][544];   // scores, then probabilities (f32)
  __shared__ alignas(16) __bf16 P[16][544];   // probabilities as WMMA A-operand
  __shared__ float red[16][16];

  const int j0   = blockIdx.x * 16;
  const int c_lo = j0 - CW_;
  const int tid  = threadIdx.x;
  const int wid  = tid >> 5;
  const int lane = tid & 31;

  // ---- Phase 1: S = q @ k^T over the band, masked ----
  v16bf qfr[4];
  for (int k = 0; k < 4; ++k)
    qfr[k] = loadA(qb + j0 * DH_ + k * 32, DH_, lane);

  for (int t = wid; t < 33; t += 8) {
    const int c0 = c_lo + t * 16;
    v8f acc = {};
    for (int k = 0; k < 4; ++k) {
      const int bn = lane & 15, bk = (lane >> 4) << 4;
      int tok = c0 + bn;                                    // per-lane clamp: masked later
      tok = tok < 0 ? 0 : (tok > N_ - 1 ? N_ - 1 : tok);
      v16bf b = *(const v16bf*)(kb + tok * DH_ + k * 32 + bk);
      acc = wmma_bf16(qfr[k], b, acc);
    }
    const int cn = lane & 15, mh = (lane >> 4) * 8;
    for (int r = 0; r < 8; ++r) {
      const int m = mh + r, j = j0 + m, c = c0 + cn;
      const bool ok = (c >= 0) & (c < N_) & (c >= j - CW_) & (c < j + CW_);
      S[m][t * 16 + cn] = ok ? acc[r] : NEGBIG;
    }
  }
  __syncthreads();

  // ---- Phase 2: softmax, 16 threads per row, 34 columns each ----
  const int row = tid >> 4, part = tid & 15;
  float mx = NEGBIG;
  for (int i = 0; i < 34; ++i) {
    const int c = part * 34 + i;
    if (c < 528) mx = fmaxf(mx, S[row][c]);
  }
  red[row][part] = mx;
  __syncthreads();
  float rm = red[row][0];
  for (int i = 1; i < 16; ++i) rm = fmaxf(rm, red[row][i]);
  __syncthreads();
  float s = 0.f;
  for (int i = 0; i < 34; ++i) {
    const int c = part * 34 + i;
    const float p = (c < 528) ? __expf(S[row][c] - rm) : 0.f;
    S[row][c] = p;
    s += p;
  }
  red[row][part] = s;
  __syncthreads();
  float rs = 0.f;
  for (int i = 0; i < 16; ++i) rs += red[row][i];
  const float inv = 1.0f / rs;
  for (int i = 0; i < 34; ++i) {
    const int c = part * 34 + i;
    P[row][c] = f2bf(S[row][c] * inv);
  }
  __syncthreads();

  // ---- Phase 3: O = P @ Vd  (K = 544, 17 WMMA steps; wave owns 16 dh cols) ----
  const int n0 = wid * 16;
  v8f acc = {};
  for (int kk0 = 0; kk0 < 544; kk0 += 32) {
    v16bf a = loadA(&P[0][kk0], 544, lane);                 // ds_load_b128 from LDS
    const int bn = lane & 15, bk = (lane >> 4) << 4;
    const int dh = n0 + bn;
    int ts = c_lo + kk0 + bk;                               // 16-token group start
    ts = ts < 0 ? 0 : (ts > N_ - 16 ? N_ - 16 : ts);        // OOB rows have P == 0
    v16bf b = *(const v16bf*)(vdT + dh * N_ + ts);          // vdT[dh][tok] contiguous
    acc = wmma_bf16(a, b, acc);
  }
  const int cn = lane & 15, mh = (lane >> 4) * 8;
  for (int r = 0; r < 8; ++r)
    attn[(j0 + mh + r) * DH_ + n0 + cn] = f2bf(acc[r]);
}

// ---------------------------------------------------------------------------
// delta = O @ Wvu^T; out = e + delta; out /= mean(out); (out-m)/sd + m; e += result
// Wave holds its 8 output tiles in registers; A fragments cached (loaded once).
// ---------------------------------------------------------------------------
__global__ void __launch_bounds__(256) k_delta_norm(const __bf16* __restrict__ attn,
                                                    const __bf16* __restrict__ wvu,
                                                    float* __restrict__ e) {
  __shared__ float red1[16][128];
  __shared__ float red2[16][128];
  __shared__ float s_invmean[16], s_m[16], s_isd[16];

  const int j0   = blockIdx.x * 16;
  const int tid  = threadIdx.x;
  const int wid  = tid >> 5;
  const int lane = tid & 31;
  const int n = lane & 15, mh = (lane >> 4) * 8;

  // Phase A: out0 = e + O @ Wvu^T (8 tiles of 16 cols per wave, K = 128)
  v16bf afr[4];
  for (int k = 0; k < 4; ++k)
    afr[k] = loadA(attn + j0 * DH_ + k * 32, DH_, lane);

  v8f out0[8];
  for (int i = 0; i < 8; ++i) {
    const int n0 = (wid * 8 + i) * 16;
    v8f acc = {};
    for (int k = 0; k < 4; ++k) {
      v16bf b = loadB(wvu + n0 * DH_ + k * 32, DH_, lane);  // Wvu[dcol][dh] contig in dh
      acc = wmma_bf16(afr[k], b, acc);
    }
    for (int r = 0; r < 8; ++r)
      acc[r] += e[(j0 + mh + r) * D_ + n0 + n];
    out0[i] = acc;
  }

  // Row sums of out0 -> inv_mean
  for (int r = 0; r < 8; ++r) {
    float s = 0.f;
    for (int i = 0; i < 8; ++i) s += out0[i][r];
    red1[mh + r][wid * 16 + n] = s;
  }
  __syncthreads();
  if (tid < 16) {
    float s = 0.f;
    for (int i = 0; i < 128; ++i) s += red1[tid][i];
    s_invmean[tid] = 1024.0f / s;                            // 1/mean
  }
  __syncthreads();

  // y = out0/mean; row mean + sumsq of y for std (ddof=1)
  for (int r = 0; r < 8; ++r) {
    const float im = s_invmean[mh + r];
    float sy = 0.f, sq = 0.f;
    for (int i = 0; i < 8; ++i) {
      const float y = out0[i][r] * im;
      sy += y; sq += y * y;
    }
    red1[mh + r][wid * 16 + n] = sy;
    red2[mh + r][wid * 16 + n] = sq;
  }
  __syncthreads();
  if (tid < 16) {
    float sy = 0.f, sq = 0.f;
    for (int i = 0; i < 128; ++i) { sy += red1[tid][i]; sq += red2[tid][i]; }
    const float m   = sy * (1.0f / 1024.0f);
    const float var = (sq - 1024.0f * m * m) * (1.0f / 1023.0f);
    s_m[tid]   = m;
    s_isd[tid] = rsqrtf(var);
  }
  __syncthreads();

  // e += (y - m)*inv_sd + m
  for (int i = 0; i < 8; ++i) {
    const int n0 = (wid * 8 + i) * 16;
    for (int r = 0; r < 8; ++r) {
      const int row = mh + r;
      const float y   = out0[i][r] * s_invmean[row];
      const float res = (y - s_m[row]) * s_isd[row] + s_m[row];
      const int idx = (j0 + row) * D_ + n0 + n;
      e[idx] += res;                                         // same-thread RMW only
    }
  }
}

__global__ void __launch_bounds__(256) k_scale(float* __restrict__ e) {
  const int i = (blockIdx.x * 256 + threadIdx.x) * 4;
  float4 v = *(float4*)(e + i);
  v.x *= 0.125f; v.y *= 0.125f; v.z *= 0.125f; v.w *= 0.125f;
  *(float4*)(e + i) = v;
}

// ---------------------------------------------------------------------------
extern "C" void kernel_launch(void* const* d_in, const int* in_sizes, int n_in,
                              void* d_out, int out_size, void* d_ws, size_t ws_size,
                              hipStream_t stream) {
  const float* x   = (const float*)d_in[0];
  const float* Wq  = (const float*)d_in[1];
  const float* Wk  = (const float*)d_in[2];
  const float* Wvd = (const float*)d_in[3];
  const float* Wvu = (const float*)d_in[4];
  float* e = (float*)d_out;                                  // e lives in d_out

  char* ws = (char*)d_ws;
  size_t o = 0;
  auto take = [&](size_t bytes) -> char* {
    char* p = ws + o;
    o += (bytes + 255) & ~(size_t)255;
    return p;
  };
  __bf16* ebf  = (__bf16*)take((size_t)N_ * D_ * 2);         // 8 MB
  __bf16* qb   = (__bf16*)take((size_t)N_ * DH_ * 2);        // 1 MB
  __bf16* kb   = (__bf16*)take((size_t)N_ * DH_ * 2);        // 1 MB
  __bf16* vdT  = (__bf16*)take((size_t)DH_ * N_ * 2);        // 1 MB (transposed V)
  __bf16* ab   = (__bf16*)take((size_t)N_ * DH_ * 2);        // 1 MB (attention out)
  __bf16* wqb  = (__bf16*)take((size_t)H_ * DH_ * D_ * 2);   // 2 MB
  __bf16* wkb  = (__bf16*)take((size_t)H_ * DH_ * D_ * 2);
  __bf16* wvdb = (__bf16*)take((size_t)H_ * DH_ * D_ * 2);
  __bf16* wvub = (__bf16*)take((size_t)H_ * D_ * DH_ * 2);

  const int WN = H_ * DH_ * D_;                              // 1,048,576
  k_f32_to_bf16<<<WN / 1024, 256, 0, stream>>>(Wq,  wqb);
  k_f32_to_bf16<<<WN / 1024, 256, 0, stream>>>(Wk,  wkb);
  k_f32_to_bf16<<<WN / 1024, 256, 0, stream>>>(Wvd, wvdb);
  k_f32_to_bf16<<<WN / 1024, 256, 0, stream>>>(Wvu, wvub);

  hipMemcpyAsync(e, x, (size_t)N_ * D_ * sizeof(float),
                 hipMemcpyDeviceToDevice, stream);

  for (int h = 0; h < H_; ++h) {
    k_f32_to_bf16<<<(N_ * D_) / 1024, 256, 0, stream>>>(e, ebf);
    k_qkv<<<N_ / 16, 192, 0, stream>>>(ebf,
        wqb  + (size_t)h * DH_ * D_,
        wkb  + (size_t)h * DH_ * D_,
        wvdb + (size_t)h * DH_ * D_,
        qb, kb, vdT);
    k_attn<<<N_ / 16, 256, 0, stream>>>(qb, kb, vdT, ab);
    k_delta_norm<<<N_ / 16, 256, 0, stream>>>(ab, wvub + (size_t)h * D_ * DH_, e);
  }
  k_scale<<<(N_ * D_) / 1024, 256, 0, stream>>>(e);

  (void)in_sizes; (void)n_in; (void)out_size; (void)ws_size;
}
